// tsp_coder_attention_56495999811907
// MI455X (gfx1250) — matrix operand order repaired and checked by reading, not compiled, not run on hardware
//
#include <hip/hip_runtime.h>
#include <math.h>

// ---------------- problem constants ----------------
constexpr int Bb  = 4;
constexpr int Nn  = 1000;
constexpr int Kk  = 16;
constexpr int Tt  = 16;
constexpr int NKe = Nn * Kk;      // 16000 edges per batch
constexpr int BNK = Bb * NKe;     // 64000 edges total

typedef _Float16 v8h  __attribute__((ext_vector_type(8)));
typedef _Float16 v16h __attribute__((ext_vector_type(16)));
typedef float    v8f  __attribute__((ext_vector_type(8)));

__device__ __forceinline__ float relu_(float x) { return fmaxf(x, 0.f); }

// =====================================================================
// edge-type net: h = relu(W1*[dx,dy,pw]+b1); softmax(W2*h+b2) over T=16
// one thread per edge, weights staged in LDS
// =====================================================================
__global__ __launch_bounds__(128) void k_etype(
    const float* __restrict__ pts, const float* __restrict__ pw,
    const int* __restrict__ nn, const float* __restrict__ e1W,
    const float* __restrict__ e1b, const float* __restrict__ e2W,
    const float* __restrict__ e2b, float* __restrict__ etype)
{
    __shared__ float sW1[64 * 3], sB1[64], sW2[16 * 64], sB2[16];
    int tid = threadIdx.x;
    for (int i = tid; i < 64 * 3; i += 128) sW1[i] = e1W[i];
    for (int i = tid; i < 64; i += 128)     sB1[i] = e1b[i];
    for (int i = tid; i < 16 * 64; i += 128) sW2[i] = e2W[i];
    if (tid < 16) sB2[tid] = e2b[tid];
    __syncthreads();
    int gid = blockIdx.x * 128 + tid;
    if (gid >= BNK) return;
    int b = gid / NKe, r = gid % NKe, n = r / Kk;
    int idx = nn[gid];
    float dx = pts[(b * 2 + 0) * Nn + idx] - pts[(b * 2 + 0) * Nn + n];
    float dy = pts[(b * 2 + 1) * Nn + idx] - pts[(b * 2 + 1) * Nn + n];
    float w  = pw[gid];
    float h[64];
#pragma unroll
    for (int j = 0; j < 64; j++)
        h[j] = relu_(sW1[j * 3] * dx + sW1[j * 3 + 1] * dy + sW1[j * 3 + 2] * w + sB1[j]);
    float lg[16];
    float m = -1e30f;
#pragma unroll
    for (int t = 0; t < 16; t++) {
        float s = sB2[t];
#pragma unroll
        for (int j = 0; j < 64; j++) s += sW2[t * 64 + j] * h[j];
        lg[t] = s; m = fmaxf(m, s);
    }
    float ssum = 0.f;
#pragma unroll
    for (int t = 0; t < 16; t++) { lg[t] = __expf(lg[t] - m); ssum += lg[t]; }
    float inv = 1.f / ssum;
#pragma unroll
    for (int t = 0; t < 16; t++) etype[(b * Tt + t) * NKe + r] = lg[t] * inv;
}

// =====================================================================
// convert W_type [T][NOUT][NIN] fp32 -> A-major f16 [NOUT][T*NIN]
// =====================================================================
__global__ void k_cvt_wtype(const float* __restrict__ Wt, _Float16* __restrict__ out,
                            int NOUT, int NIN)
{
    int gid = blockIdx.x * 256 + threadIdx.x;
    int total = Tt * NOUT * NIN;
    if (gid >= total) return;
    int c = gid % NIN, o = (gid / NIN) % NOUT, t = gid / (NIN * NOUT);
    out[o * (Tt * NIN) + t * NIN + c] = (_Float16)Wt[gid];
}

// =====================================================================
// WMMA fragment loaders (ISA 7.12.2 16-bit A 16x32 / B 32x16 lane layouts)
// =====================================================================
__device__ __forceinline__ v16h load_a16(const _Float16* __restrict__ p, int lane)
{
    // p = row start + colbase; lane M = lane%16;
    // lanes 0-15: K = {0..7, 16..23}; lanes 16-31: K = {8..15, 24..31}
    int koff = (lane < 16) ? 0 : 8;
    v8h lo = *(const v8h*)(p + koff);
    v8h hi = *(const v8h*)(p + koff + 16);
    v16h a;
#pragma unroll
    for (int j = 0; j < 8; j++) { a[j] = lo[j]; a[j + 8] = hi[j]; }
    return a;
}
__device__ __forceinline__ v16h load_b16(const _Float16* base, int lane)
{
    // base = &x[col = lane%16][c0]; lanes 0-15 K={0..7,16..23}, lanes 16-31 K={8..15,24..31}
    const _Float16* p = base + ((lane < 16) ? 0 : 8);
    v8h lo = *(const v8h*)p;
    v8h hi = *(const v8h*)(p + 16);
    v16h b;
#pragma unroll
    for (int j = 0; j < 8; j++) { b[j] = lo[j]; b[j + 8] = hi[j]; }
    return b;
}

// =====================================================================
// fused mp_conv: one workgroup per (b, n) = one 16-edge WMMA column tile
//   msg[o,k]  = sum_{t,c} Wt[o, t*NIN+c] * (etype[t,k] * nb[c,k])   (WMMA)
//   n_out[o]  = relu(W_self*nf + b_self + mean_k msg)
//   w_out[.,k]= relu(W_w * concat(wf, msg) + b_w)
// 128 threads = 4 wave32; each wave owns NOUT/16/4 row tiles
// =====================================================================
template <int NIN, int NOUT, int WIN, int WOUT>
__global__ __launch_bounds__(128) void k_mp_fused(
    const float* __restrict__ nf, const float* __restrict__ wf,
    const float* __restrict__ etype, const int* __restrict__ nn,
    const _Float16* __restrict__ Wt, const float* __restrict__ Wself,
    const float* __restrict__ bself, const float* __restrict__ Ww,
    const float* __restrict__ bw, float* __restrict__ nout_,
    float* __restrict__ wout_)
{
    constexpr int KKc = Tt * NIN;                 // full contraction length
    constexpr int XW  = (NIN >= 32) ? NIN : KKc;  // LDS X row length
    constexpr int RT  = NOUT / 16;                // 16-row output tiles
    constexpr int RTS = RT / 4;                   // tiles per wave (4 waves)

    __shared__ float etS[Tt][16];
    __shared__ float nbS[NIN][16];
    __shared__ float nfS[NIN];
    __shared__ float wfS[WIN][16];
    __shared__ float msgS[NOUT][16];
    __shared__ alignas(32) _Float16 xS[16][XW];
    __shared__ int idxS[16];

    int tid = threadIdx.x, lane = tid & 31, wv = tid >> 5;
    int b = blockIdx.x / Nn, n = blockIdx.x % Nn;

    if (tid < 16) idxS[tid] = nn[(b * Nn + n) * Kk + tid];
    for (int e = tid; e < Tt * 16; e += 128) {
        int t = e >> 4, k = e & 15;
        etS[t][k] = etype[(b * Tt + t) * NKe + n * Kk + k];
    }
    for (int e = tid; e < NIN; e += 128) nfS[e] = nf[(b * NIN + e) * Nn + n];
    for (int e = tid; e < WIN * 16; e += 128) {
        int i = e >> 4, k = e & 15;
        wfS[i][k] = wf[(b * WIN + i) * NKe + n * Kk + k];
    }
    __syncthreads();
    for (int e = tid; e < NIN * 16; e += 128) {
        int c = e >> 4, k = e & 15;
        nbS[c][k] = nf[(b * NIN + c) * Nn + idxS[k]];
    }
    __syncthreads();

    v8f acc[RTS] = {};

    if constexpr (NIN >= 32) {
#pragma unroll 1
        for (int t = 0; t < Tt; t++) {
            __syncthreads();                         // xS consumers of prev t done
            for (int e = tid; e < 16 * NIN; e += 128) {
                int col = e / NIN, c = e % NIN;
                xS[col][c] = (_Float16)(etS[t][col] * nbS[c][col]);
            }
            __syncthreads();
#pragma unroll
            for (int rti = 0; rti < RTS; rti++) {
                int row = (wv + 4 * rti) * 16 + (lane & 15);
                const _Float16* arow = Wt + (size_t)row * KKc + t * NIN;
#pragma unroll
                for (int c0 = 0; c0 < NIN; c0 += 32) {
                    v16h A = load_a16(arow + c0, lane);
                    v16h Bf = load_b16(&xS[lane & 15][c0], lane);
                    acc[rti] = __builtin_amdgcn_wmma_f32_16x16x32_f16(
                        false, A, false, Bf, (short)0, acc[rti], false, false);
                }
            }
            if (lane == 0 && t + 1 < Tt)  // hint next weight panel into cache
                __builtin_prefetch(Wt + (size_t)(wv * 16) * KKc + (t + 1) * NIN, 0, 1);
        }
    } else {
        // tiny NIN (mp1): whole K = T*NIN = 32 in one WMMA step; RTS == 1
        for (int e = tid; e < 16 * KKc; e += 128) {
            int col = e / KKc, kk = e % KKc;
            int t = kk / NIN, c = kk % NIN;
            xS[col][kk] = (_Float16)(etS[t][col] * nbS[c][col]);
        }
        __syncthreads();
        int row = wv * 16 + (lane & 15);
        v16h A = load_a16(Wt + (size_t)row * KKc, lane);
        v16h Bf = load_b16(&xS[lane & 15][0], lane);
        acc[0] = __builtin_amdgcn_wmma_f32_16x16x32_f16(
            false, A, false, Bf, (short)0, acc[0], false, false);
    }
    __syncthreads();
    // C/D layout: VGPR r, lanes<16 -> M=r, lanes>=16 -> M=r+8; col = lane%16
#pragma unroll
    for (int rti = 0; rti < RTS; rti++) {
#pragma unroll
        for (int r = 0; r < 8; r++) {
            int o = (wv + 4 * rti) * 16 + r + ((lane >> 4) << 3);
            msgS[o][lane & 15] = acc[rti][r];
        }
    }
    __syncthreads();

    // n_out: self GEMV + mean-over-K aggregation + relu
    for (int o = tid; o < NOUT; o += 128) {
        float agg = 0.f;
#pragma unroll
        for (int k = 0; k < 16; k++) agg += msgS[o][k];
        agg *= (1.0f / Kk);
        float s = bself[o];
        for (int c = 0; c < NIN; c++) s += Wself[o * NIN + c] * nfS[c];
        nout_[(b * NOUT + o) * Nn + n] = relu_(s + agg);
    }
    // w_out: W_w * concat(wf, msg) + b_w, relu
    for (int e = tid; e < WOUT * 16; e += 128) {
        int ow = e >> 4, k = e & 15;
        const float* wr = Ww + ow * (WIN + NOUT);
        float s = bw[ow];
        for (int i = 0; i < WIN; i++)  s += wr[i] * wfS[i][k];
        for (int j = 0; j < NOUT; j++) s += wr[WIN + j] * msgS[j][k];
        wout_[(b * WOUT + ow) * NKe + n * Kk + k] = relu_(s);
    }
}

// =====================================================================
// generic conv1x1: y[b,o,p] = bias[o] + sum_c W[o,c] x[b,c,p]
// =====================================================================
__global__ void k_conv1x1(const float* __restrict__ x, const float* __restrict__ W,
                          const float* __restrict__ bias, float* __restrict__ y,
                          int CIN, int COUT, int P)
{
    int gid = blockIdx.x * 256 + threadIdx.x;
    int total = Bb * COUT * P;
    if (gid >= total) return;
    int p = gid % P, o = (gid / P) % COUT, b = gid / (P * COUT);
    float s = bias[o];
    for (int c = 0; c < CIN; c++) s += W[o * CIN + c] * x[(b * CIN + c) * P + p];
    y[gid] = s;
}

// BatchNorm batch-statistics: one block per channel (sum, sumsq over B*P)
__global__ __launch_bounds__(256) void k_bn_stats(const float* __restrict__ x,
                                                  float* __restrict__ stats, int C, int P)
{
    int c = blockIdx.x, tid = threadIdx.x;
    float s = 0.f, s2 = 0.f;
    for (int i = tid; i < Bb * P; i += 256) {
        int b = i / P, p = i % P;
        float v = x[(b * C + c) * P + p];
        s += v; s2 += v * v;
    }
    __shared__ float rs[256], rs2[256];
    rs[tid] = s; rs2[tid] = s2; __syncthreads();
    for (int off = 128; off > 0; off >>= 1) {
        if (tid < off) { rs[tid] += rs[tid + off]; rs2[tid] += rs2[tid + off]; }
        __syncthreads();
    }
    if (tid == 0) { stats[2 * c] = rs[0]; stats[2 * c + 1] = rs2[0]; }
}

__global__ void k_bn_relu(const float* __restrict__ x, float* __restrict__ y,
                          const float* __restrict__ stats, const float* __restrict__ g,
                          const float* __restrict__ be, int C, int P)
{
    int gid = blockIdx.x * 256 + threadIdx.x;
    int total = Bb * C * P;
    if (gid >= total) return;
    int c = (gid / P) % C;
    float M = (float)(Bb * P);
    float mu = stats[2 * c] / M;
    float var = stats[2 * c + 1] / M - mu * mu;
    float yv = (x[gid] - mu) * rsqrtf(var + 1e-5f) * g[c] + be[c];
    y[gid] = relu_(yv);
}

__global__ void k_add(const float* __restrict__ a, const float* __restrict__ b,
                      float* __restrict__ y, int n)
{
    int gid = blockIdx.x * 256 + threadIdx.x;
    if (gid < n) y[gid] = a[gid] + b[gid];
}

// =====================================================================
// attention decode
// =====================================================================
__global__ __launch_bounds__(128) void k_att_glbq(const float* __restrict__ f,
    const float* __restrict__ gW, const float* __restrict__ gb, float* __restrict__ glbQ)
{
    __shared__ float fm[128];
    int b = blockIdx.x, c = threadIdx.x;
    float m = -1e30f;
    for (int n = 0; n < Nn; n++) m = fmaxf(m, f[(b * 128 + c) * Nn + n]);
    fm[c] = m; __syncthreads();
    float s = gb[c];
    for (int j = 0; j < 128; j++) s += gW[c * 128 + j] * fm[j];
    glbQ[b * 128 + c] = s;
}

__global__ __launch_bounds__(128) void k_ctxq(const float* __restrict__ f,
    const int* __restrict__ lc, const float* __restrict__ lW, const float* __restrict__ lb,
    const float* __restrict__ glbQ, float* __restrict__ ctxQ)
{
    int b = blockIdx.x, o = threadIdx.x;
    int li = lc[b * 2 + 0], ci = lc[b * 2 + 1];
    float s = lb[o] + glbQ[b * 128 + o];
    for (int c = 0; c < 128; c++) {
        s += lW[o * 256 + c] * f[(b * 128 + c) * Nn + li];
        s += lW[o * 256 + 128 + c] * f[(b * 128 + c) * Nn + ci];
    }
    ctxQ[b * 128 + o] = s;
}

__global__ __launch_bounds__(256) void k_att_dec(const float* __restrict__ kvl,
    const float* __restrict__ ctxQ, const float* __restrict__ pW,
    const float* __restrict__ pb, float* __restrict__ out)
{
    __shared__ float qS[128], sc[Nn], vS[128], pS[128], red[256];
    __shared__ float mS, sumS;
    int b = blockIdx.x, tid = threadIdx.x;
    const float scale = 0.08838834764831845f;  // 1/sqrt(128)
    if (tid < 128) qS[tid] = ctxQ[b * 128 + tid];
    __syncthreads();
    for (int n = tid; n < Nn; n += 256) {
        float s = 0.f;
        for (int c = 0; c < 128; c++) s += qS[c] * kvl[(b * 384 + c) * Nn + n];
        sc[n] = s * scale;
    }
    __syncthreads();
    float lm = -1e30f;
    for (int n = tid; n < Nn; n += 256) lm = fmaxf(lm, sc[n]);
    red[tid] = lm; __syncthreads();
    for (int off = 128; off > 0; off >>= 1) {
        if (tid < off) red[tid] = fmaxf(red[tid], red[tid + off]);
        __syncthreads();
    }
    if (tid == 0) mS = red[0];
    __syncthreads();
    float ls = 0.f;
    for (int n = tid; n < Nn; n += 256) { float e = __expf(sc[n] - mS); sc[n] = e; ls += e; }
    red[tid] = ls; __syncthreads();
    for (int off = 128; off > 0; off >>= 1) {
        if (tid < off) red[tid] += red[tid + off];
        __syncthreads();
    }
    if (tid == 0) sumS = red[0];
    __syncthreads();
    if (tid < 128) {
        int c = tid; float s = 0.f;
        for (int n = 0; n < Nn; n++) s += sc[n] * kvl[(b * 384 + 128 + c) * Nn + n];
        vS[c] = s / sumS;
    }
    __syncthreads();
    if (tid < 128) {
        int o = tid; float s = pb[o];
        for (int c = 0; c < 128; c++) s += pW[o * 128 + c] * vS[c];
        pS[o] = s;
    }
    __syncthreads();
    for (int n = tid; n < Nn; n += 256) {
        float s = 0.f;
        for (int c = 0; c < 128; c++) s += pS[c] * kvl[(b * 384 + 256 + c) * Nn + n];
        out[b * Nn + n] = tanhf(s * scale) * 10.f;
    }
}

// =====================================================================
// host orchestration
// =====================================================================
static inline unsigned g256(long n) { return (unsigned)((n + 255) / 256); }

extern "C" void kernel_launch(void* const* d_in, const int* in_sizes, int n_in,
                              void* d_out, int out_size, void* d_ws, size_t ws_size,
                              hipStream_t stream)
{
    (void)in_sizes; (void)n_in; (void)out_size; (void)ws_size;
    int i = 0;
    const float* pts = (const float*)d_in[i++];          // [B,2,N]
    const float* pw  = (const float*)d_in[i++];          // [B,1,N,K]
    const int*   nn  = (const int*)d_in[i++];            // [B,N,K]
    const int*   lc  = (const int*)d_in[i++];            // [B,2]
    // params pytree in JAX flatten order (dict keys sorted at each level):
    // att: glb_W, glb_b, lc_W, lc_b, node_W, node_b, proj_W, proj_b
    const float* att_glbW  = (const float*)d_in[i++]; const float* att_glbB  = (const float*)d_in[i++];
    const float* att_lcW   = (const float*)d_in[i++]; const float* att_lcB   = (const float*)d_in[i++];
    const float* att_nodeW = (const float*)d_in[i++]; const float* att_nodeB = (const float*)d_in[i++];
    const float* att_projW = (const float*)d_in[i++]; const float* att_projB = (const float*)d_in[i++];
    // etype: e1_W, e1_b, e2_W, e2_b
    const float* e1W = (const float*)d_in[i++]; const float* e1b = (const float*)d_in[i++];
    const float* e2W = (const float*)d_in[i++]; const float* e2b = (const float*)d_in[i++];
    // mp: W_self, W_type, W_w, b_self, b_w
    const float* mp1_Ws = (const float*)d_in[i++]; const float* mp1_Wt = (const float*)d_in[i++];
    const float* mp1_Ww = (const float*)d_in[i++]; const float* mp1_bs = (const float*)d_in[i++];
    const float* mp1_bw = (const float*)d_in[i++];
    const float* mp2_Ws = (const float*)d_in[i++]; const float* mp2_Wt = (const float*)d_in[i++];
    const float* mp2_Ww = (const float*)d_in[i++]; const float* mp2_bs = (const float*)d_in[i++];
    const float* mp2_bw = (const float*)d_in[i++];
    // res: mp{...}, then n1_W,n1_b,n1_be,n1_g, n2_*, w1_*, w2_*
    const float* r1_Ws = (const float*)d_in[i++]; const float* r1_Wt = (const float*)d_in[i++];
    const float* r1_Ww = (const float*)d_in[i++]; const float* r1_bs = (const float*)d_in[i++];
    const float* r1_bw = (const float*)d_in[i++];
    const float* r1_n1W = (const float*)d_in[i++]; const float* r1_n1b = (const float*)d_in[i++];
    const float* r1_n1be = (const float*)d_in[i++]; const float* r1_n1g = (const float*)d_in[i++];
    const float* r1_n2W = (const float*)d_in[i++]; const float* r1_n2b = (const float*)d_in[i++];
    const float* r1_n2be = (const float*)d_in[i++]; const float* r1_n2g = (const float*)d_in[i++];
    const float* r1_w1W = (const float*)d_in[i++]; const float* r1_w1b = (const float*)d_in[i++];
    const float* r1_w1be = (const float*)d_in[i++]; const float* r1_w1g = (const float*)d_in[i++];
    const float* r1_w2W = (const float*)d_in[i++]; const float* r1_w2b = (const float*)d_in[i++];
    const float* r1_w2be = (const float*)d_in[i++]; const float* r1_w2g = (const float*)d_in[i++];
    const float* r2_Ws = (const float*)d_in[i++]; const float* r2_Wt = (const float*)d_in[i++];
    const float* r2_Ww = (const float*)d_in[i++]; const float* r2_bs = (const float*)d_in[i++];
    const float* r2_bw = (const float*)d_in[i++];
    const float* r2_n1W = (const float*)d_in[i++]; const float* r2_n1b = (const float*)d_in[i++];
    const float* r2_n1be = (const float*)d_in[i++]; const float* r2_n1g = (const float*)d_in[i++];
    const float* r2_n2W = (const float*)d_in[i++]; const float* r2_n2b = (const float*)d_in[i++];
    const float* r2_n2be = (const float*)d_in[i++]; const float* r2_n2g = (const float*)d_in[i++];
    const float* r2_w1W = (const float*)d_in[i++]; const float* r2_w1b = (const float*)d_in[i++];
    const float* r2_w1be = (const float*)d_in[i++]; const float* r2_w1g = (const float*)d_in[i++];
    (void)d_in[i]; // r2_w2_* (dead branch: res2 w-output never feeds the decoder)

    // ---- workspace bump allocator ----
    char* wsp = (char*)d_ws;
    auto alloc = [&](size_t bytes) -> void* {
        void* p = (void*)wsp; wsp += (bytes + 255) & ~size_t(255); return p;
    };
    float* etype = (float*)alloc(sizeof(float) * Bb * Tt * NKe);
    _Float16* Wt1h  = (_Float16*)alloc(sizeof(_Float16) * 64 * 32);
    _Float16* Wtr1h = (_Float16*)alloc(sizeof(_Float16) * 64 * 1024);
    _Float16* Wt2h  = (_Float16*)alloc(sizeof(_Float16) * 128 * 1024);
    _Float16* Wtr2h = (_Float16*)alloc(sizeof(_Float16) * 128 * 2048);
    size_t nbytes = sizeof(float) * Bb * 128 * Nn;
    size_t wbytes = sizeof(float) * Bb * 32 * NKe;
    float* nA = (float*)alloc(nbytes); float* nB = (float*)alloc(nbytes);
    float* nC = (float*)alloc(nbytes); float* nD = (float*)alloc(nbytes);
    float* wA = (float*)alloc(wbytes); float* wB = (float*)alloc(wbytes);
    float* wC = (float*)alloc(wbytes); float* wD = (float*)alloc(wbytes);
    float* stats = (float*)alloc(sizeof(float) * 256);
    float* glbQ = (float*)alloc(sizeof(float) * Bb * 128);
    float* ctxQ = (float*)alloc(sizeof(float) * Bb * 128);
    float* kvl  = (float*)alloc(sizeof(float) * Bb * 384 * Nn);

    // ---- edge-type net ----
    k_etype<<<(BNK + 127) / 128, 128, 0, stream>>>(pts, pw, nn, e1W, e1b, e2W, e2b, etype);

    // ---- mp1 ----
    k_cvt_wtype<<<g256(16 * 64 * 2), 256, 0, stream>>>(mp1_Wt, Wt1h, 64, 2);
    k_mp_fused<2, 64, 1, 16><<<Bb * Nn, 128, 0, stream>>>(
        pts, pw, etype, nn, Wt1h, mp1_Ws, mp1_bs, mp1_Ww, mp1_bw, nA, wA);

    // ---- res1 ----
    k_conv1x1<<<g256((long)Bb * 64 * Nn), 256, 0, stream>>>(nA, r1_n1W, r1_n1b, nB, 64, 64, Nn);
    k_bn_stats<<<64, 256, 0, stream>>>(nB, stats, 64, Nn);
    k_bn_relu<<<g256((long)Bb * 64 * Nn), 256, 0, stream>>>(nB, nB, stats, r1_n1g, r1_n1be, 64, Nn);
    k_conv1x1<<<g256((long)Bb * 16 * NKe), 256, 0, stream>>>(wA, r1_w1W, r1_w1b, wB, 16, 16, NKe);
    k_bn_stats<<<16, 256, 0, stream>>>(wB, stats, 16, NKe);
    k_bn_relu<<<g256((long)Bb * 16 * NKe), 256, 0, stream>>>(wB, wB, stats, r1_w1g, r1_w1be, 16, NKe);
    k_cvt_wtype<<<g256(16 * 64 * 64), 256, 0, stream>>>(r1_Wt, Wtr1h, 64, 64);
    k_mp_fused<64, 64, 16, 16><<<Bb * Nn, 128, 0, stream>>>(
        nB, wB, etype, nn, Wtr1h, r1_Ws, r1_bs, r1_Ww, r1_bw, nC, wC);
    k_conv1x1<<<g256((long)Bb * 64 * Nn), 256, 0, stream>>>(nC, r1_n2W, r1_n2b, nD, 64, 64, Nn);
    k_bn_stats<<<64, 256, 0, stream>>>(nD, stats, 64, Nn);
    k_bn_relu<<<g256((long)Bb * 64 * Nn), 256, 0, stream>>>(nD, nD, stats, r1_n2g, r1_n2be, 64, Nn);
    k_conv1x1<<<g256((long)Bb * 16 * NKe), 256, 0, stream>>>(wC, r1_w2W, r1_w2b, wD, 16, 16, NKe);
    k_bn_stats<<<16, 256, 0, stream>>>(wD, stats, 16, NKe);
    k_bn_relu<<<g256((long)Bb * 16 * NKe), 256, 0, stream>>>(wD, wD, stats, r1_w2g, r1_w2be, 16, NKe);
    k_add<<<g256((long)Bb * 64 * Nn), 256, 0, stream>>>(nD, nA, nD, Bb * 64 * Nn);
    k_add<<<g256((long)Bb * 16 * NKe), 256, 0, stream>>>(wD, wA, wD, Bb * 16 * NKe);

    // ---- mp2 ----
    k_cvt_wtype<<<g256(16 * 128 * 64), 256, 0, stream>>>(mp2_Wt, Wt2h, 128, 64);
    k_mp_fused<64, 128, 16, 32><<<Bb * Nn, 128, 0, stream>>>(
        nD, wD, etype, nn, Wt2h, mp2_Ws, mp2_bs, mp2_Ww, mp2_bw, nA, wA);

    // ---- res2 (dead w2 branch eliminated) ----
    k_conv1x1<<<g256((long)Bb * 128 * Nn), 256, 0, stream>>>(nA, r2_n1W, r2_n1b, nB, 128, 128, Nn);
    k_bn_stats<<<128, 256, 0, stream>>>(nB, stats, 128, Nn);
    k_bn_relu<<<g256((long)Bb * 128 * Nn), 256, 0, stream>>>(nB, nB, stats, r2_n1g, r2_n1be, 128, Nn);
    k_conv1x1<<<g256((long)Bb * 32 * NKe), 256, 0, stream>>>(wA, r2_w1W, r2_w1b, wB, 32, 32, NKe);
    k_bn_stats<<<32, 256, 0, stream>>>(wB, stats, 32, NKe);
    k_bn_relu<<<g256((long)Bb * 32 * NKe), 256, 0, stream>>>(wB, wB, stats, r2_w1g, r2_w1be, 32, NKe);
    k_cvt_wtype<<<g256(16 * 128 * 128), 256, 0, stream>>>(r2_Wt, Wtr2h, 128, 128);
    k_mp_fused<128, 128, 32, 32><<<Bb * Nn, 128, 0, stream>>>(
        nB, wB, etype, nn, Wtr2h, r2_Ws, r2_bs, r2_Ww, r2_bw, nC, wC);
    k_conv1x1<<<g256((long)Bb * 128 * Nn), 256, 0, stream>>>(nC, r2_n2W, r2_n2b, nD, 128, 128, Nn);
    k_bn_stats<<<128, 256, 0, stream>>>(nD, stats, 128, Nn);
    k_bn_relu<<<g256((long)Bb * 128 * Nn), 256, 0, stream>>>(nD, nD, stats, r2_n2g, r2_n2be, 128, Nn);
    k_add<<<g256((long)Bb * 128 * Nn), 256, 0, stream>>>(nD, nA, nD, Bb * 128 * Nn);

    // ---- attention decode ----
    k_att_glbq<<<Bb, 128, 0, stream>>>(nD, att_glbW, att_glbB, glbQ);
    k_conv1x1<<<g256((long)Bb * 384 * Nn), 256, 0, stream>>>(nD, att_nodeW, att_nodeB, kvl, 128, 384, Nn);
    k_ctxq<<<Bb, 128, 0, stream>>>(nD, lc, att_lcW, att_lcB, glbQ, ctxQ);
    k_att_dec<<<Bb, 256, 0, stream>>>(kvl, ctxQ, att_projW, att_projB, (float*)d_out);
}
